// ConcatAttention_38792144618060
// MI455X (gfx1250) — matrix operand order repaired and checked by reading, not compiled, not run on hardware
//
#include <hip/hip_runtime.h>

// ---------------------------------------------------------------------------
// Additive (Bahdanau) attention, MI455X / gfx1250.
// Dominant cost: all_encs[16,8192,512] @ W2[512,512]  (~68.7 GFLOP) -> do it
// with v_wmma_f32_16x16x32_bf16 (fp32 accumulate), A converted to bf16 while
// staging to LDS, W2 converted/transposed to bf16 once in workspace.
// ---------------------------------------------------------------------------

typedef __bf16 bf16_t;
typedef bf16_t v16bf __attribute__((ext_vector_type(16)));
typedef bf16_t v8bf  __attribute__((ext_vector_type(8)));
typedef float  v8f   __attribute__((ext_vector_type(8)));

#define B_ 16
#define S_ 8192
#define E_ 512
#define U_ 512

#define SROWS    64                 // S-rows per block in the score kernel
#define LDPAD    8                  // bf16 elems of padding per LDS row
#define LDSTRIDE (E_ + LDPAD)       // 520 bf16 = 1040 B (65*16: keeps 16B align)
#define SCHUNK   256                // S-rows per block in context partial pass
#define NCHUNK   (S_ / SCHUNK)      // 32

#define FLT_MAX_ 3.402823466e+38f

// ---------------------------------------------------------------------------
// Kernel 0: W2T[u][e] = bf16(W2[e][u])   (512 KB in ws; B frags then contiguous)
// ---------------------------------------------------------------------------
__global__ __launch_bounds__(256) void k_w2t(const float* __restrict__ W2,
                                             bf16_t* __restrict__ W2T) {
  int idx = blockIdx.x * 256 + threadIdx.x;   // 0 .. 262143
  int e = idx >> 9;
  int u = idx & (U_ - 1);
  W2T[(size_t)u * E_ + e] = (bf16_t)W2[(size_t)e * U_ + u];
}

// ---------------------------------------------------------------------------
// Kernel 1: qb[b][u] = hidden[b]·W1[:,u] + b1[u] + b2[u]   (tiny, fp32 exact)
// ---------------------------------------------------------------------------
__global__ __launch_bounds__(256) void k_qb(const float* __restrict__ hidden,
                                            const float* __restrict__ W1,
                                            const float* __restrict__ b1,
                                            const float* __restrict__ b2,
                                            float* __restrict__ qb) {
  int idx = blockIdx.x * 256 + threadIdx.x;   // 0 .. 8191
  int b = idx >> 9;
  int u = idx & (U_ - 1);
  float s = b1[u] + b2[u];
  const float* h = hidden + (size_t)b * E_;
  for (int e = 0; e < E_; ++e)
    s = fmaf(h[e], W1[(size_t)e * U_ + u], s);   // coalesced on u
  qb[idx] = s;
}

// ---------------------------------------------------------------------------
// Kernel 2: fused k-projection + tanh + V-dot -> scores[b][s]
//   grid: B * (S/64) blocks of 512 threads (16 waves).
//   Wave (mp = w&1, ng = w>>1): rows [mp*32, mp*32+32), cols [ng*64, ng*64+64)
//   -> 2 M-tiles x 4 N-tiles of 16x16, B fragments reused across M-tiles.
// ---------------------------------------------------------------------------
__global__ __launch_bounds__(512) void k_scores(const float*  __restrict__ encs,
                                                const bf16_t* __restrict__ W2T,
                                                const float*  __restrict__ qb,
                                                const float*  __restrict__ V,
                                                const float*  __restrict__ bV,
                                                float* __restrict__ scores) {
  extern __shared__ char smem[];
  bf16_t* sA = (bf16_t*)smem;                                    // [64][520] bf16
  float*  sP = (float*)(smem + SROWS * LDSTRIDE * sizeof(bf16_t)); // [8][64] f32

  const int tid  = threadIdx.x;
  const int wave = tid >> 5;
  const int lane = tid & 31;
  const int half = lane >> 4;      // 0: lanes 0-15, 1: lanes 16-31
  const int l15  = lane & 15;

  const int blocksPerBatch = S_ / SROWS;           // 128
  const int b     = blockIdx.x / blocksPerBatch;
  const int sbase = (blockIdx.x % blocksPerBatch) * SROWS;

  // ---- stage 64x512 enc tile: fp32 global -> bf16 LDS (RNE via cast) ----
  {
    const float4* src =
        (const float4*)(encs + ((size_t)b * S_ + sbase) * E_);
    for (int i = tid; i < (SROWS * E_) / 4; i += 512) {
      float4 v = src[i];
      int row = (i * 4) >> 9;
      int col = (i * 4) & (E_ - 1);
      bf16_t* dst = sA + row * LDSTRIDE + col;
      dst[0] = (bf16_t)v.x; dst[1] = (bf16_t)v.y;
      dst[2] = (bf16_t)v.z; dst[3] = (bf16_t)v.w;
    }
  }
  __syncthreads();

  const int mp    = wave & 1;       // M-pair: rows mp*32 .. mp*32+31
  const int ng    = wave >> 1;      // N-group: cols ng*64 .. ng*64+63
  const int nbase = ng * 64;

  v8f acc[2][4];
#pragma unroll
  for (int m = 0; m < 2; ++m)
#pragma unroll
    for (int n = 0; n < 4; ++n)
#pragma unroll
      for (int r = 0; r < 8; ++r) acc[m][n][r] = 0.0f;

  // A layout (16-bit 16x32): lanes 0-15 hold M=l15, K in {0..7, 16..23};
  //                          lanes 16-31 hold M=l15, K in {8..15, 24..31}.
  const int rowA0 = mp * 32 + l15;
  const int rowA1 = rowA0 + 16;
  const int ksel  = half * 8;

  for (int kk = 0; kk < E_; kk += 32) {
    // --- A fragments from LDS (16B ds loads, padded rows -> no bank conflict)
    v8bf a0lo = *(const v8bf*)(sA + rowA0 * LDSTRIDE + kk + ksel);
    v8bf a0hi = *(const v8bf*)(sA + rowA0 * LDSTRIDE + kk + 16 + ksel);
    v8bf a1lo = *(const v8bf*)(sA + rowA1 * LDSTRIDE + kk + ksel);
    v8bf a1hi = *(const v8bf*)(sA + rowA1 * LDSTRIDE + kk + 16 + ksel);
    v16bf a0 = __builtin_shufflevector(a0lo, a0hi, 0,1,2,3,4,5,6,7,8,9,10,11,12,13,14,15);
    v16bf a1 = __builtin_shufflevector(a1lo, a1hi, 0,1,2,3,4,5,6,7,8,9,10,11,12,13,14,15);

    // --- B fragments (32x16): lanes 0-15 -> N=l15, K in {0..7, 16..23};
    //                          lanes 16-31 -> N=l15, K in {8..15, 24..31}.
    v16bf bfr[4];
#pragma unroll
    for (int n = 0; n < 4; ++n) {
      int u = nbase + n * 16 + l15;
      v8bf blo = *(const v8bf*)(W2T + (size_t)u * E_ + kk + ksel);
      v8bf bhi = *(const v8bf*)(W2T + (size_t)u * E_ + kk + 16 + ksel);
      bfr[n] = __builtin_shufflevector(blo, bhi, 0,1,2,3,4,5,6,7,8,9,10,11,12,13,14,15);
    }

    // --- 8 WMMAs: 131072 FLOP per wave per iteration
#pragma unroll
    for (int n = 0; n < 4; ++n) {
      acc[0][n] = __builtin_amdgcn_wmma_f32_16x16x32_bf16(
          false, a0, false, bfr[n], (short)0, acc[0][n], false, false);
      acc[1][n] = __builtin_amdgcn_wmma_f32_16x16x32_bf16(
          false, a1, false, bfr[n], (short)0, acc[1][n], false, false);
    }
  }

  // ---- fused epilogue: score_row += sum_u tanh(k + qb) * V[u] ----
  // C/D layout: VGPR r -> M = r + 8*half, N = l15 (within each 16x16 tile).
  float part[2][8];
#pragma unroll
  for (int m = 0; m < 2; ++m)
#pragma unroll
    for (int r = 0; r < 8; ++r) part[m][r] = 0.0f;

#pragma unroll
  for (int n = 0; n < 4; ++n) {
    int u = nbase + n * 16 + l15;
    float qv = qb[(size_t)b * U_ + u];
    float vv = V[u];
#pragma unroll
    for (int m = 0; m < 2; ++m)
#pragma unroll
      for (int r = 0; r < 8; ++r)
        part[m][r] += tanhf(acc[m][n][r] + qv) * vv;
  }

  // reduce the 16 lanes sharing each row set (xor masks < 16 keep halves apart)
#pragma unroll
  for (int m = 0; m < 2; ++m)
#pragma unroll
    for (int r = 0; r < 8; ++r) {
      float p = part[m][r];
      p += __shfl_xor(p, 1, 32);
      p += __shfl_xor(p, 2, 32);
      p += __shfl_xor(p, 4, 32);
      p += __shfl_xor(p, 8, 32);
      part[m][r] = p;
    }

  // deterministic per-ng partial store, then fixed-order reduction
  if (l15 == 0) {
#pragma unroll
    for (int m = 0; m < 2; ++m)
#pragma unroll
      for (int r = 0; r < 8; ++r)
        sP[ng * SROWS + (mp * 32 + m * 16 + half * 8 + r)] = part[m][r];
  }
  __syncthreads();
  if (tid < SROWS) {
    float s = bV[0];
#pragma unroll
    for (int g = 0; g < 8; ++g) s += sP[g * SROWS + tid];
    scores[(size_t)b * S_ + sbase + tid] = s;
  }
}

// ---------------------------------------------------------------------------
// Kernel 3: masked softmax over S per batch -> alignment (d_out[8192:])
// ---------------------------------------------------------------------------
__global__ __launch_bounds__(1024) void k_softmax(const float* __restrict__ scores,
                                                  const int* __restrict__ mask_step,
                                                  float* __restrict__ out) {
  const int b   = blockIdx.x;
  const int tid = threadIdx.x;
  const int ms  = mask_step[0];
  float* align  = out + B_ * E_;

  __shared__ float red[1024];

  float vals[8];
  float vmax = -FLT_MAX_;
#pragma unroll
  for (int i = 0; i < 8; ++i) {
    int s = tid + i * 1024;
    float sc = scores[(size_t)b * S_ + s];
    if (s == ms) sc = -FLT_MAX_;            // finfo(f32).min, matches reference
    vals[i] = sc;
    vmax = fmaxf(vmax, sc);
  }
  red[tid] = vmax; __syncthreads();
  for (int off = 512; off > 0; off >>= 1) {
    if (tid < off) red[tid] = fmaxf(red[tid], red[tid + off]);
    __syncthreads();
  }
  vmax = red[0]; __syncthreads();

  float vsum = 0.0f;
#pragma unroll
  for (int i = 0; i < 8; ++i) { vals[i] = __expf(vals[i] - vmax); vsum += vals[i]; }
  red[tid] = vsum; __syncthreads();
  for (int off = 512; off > 0; off >>= 1) {
    if (tid < off) red[tid] += red[tid + off];
    __syncthreads();
  }
  float inv = 1.0f / red[0];
#pragma unroll
  for (int i = 0; i < 8; ++i)
    align[(size_t)b * S_ + tid + i * 1024] = vals[i] * inv;
}

// ---------------------------------------------------------------------------
// Kernel 4a/4b: context = alignment @ all_encs   (memory-bound second pass
// over all_encs; deterministic two-stage reduction through workspace).
// ---------------------------------------------------------------------------
__global__ __launch_bounds__(256) void k_context_part(const float* __restrict__ encs,
                                                      const float* __restrict__ align,
                                                      float* __restrict__ part) {
  const int b   = blockIdx.x / NCHUNK;
  const int c   = blockIdx.x % NCHUNK;
  const int s0  = c * SCHUNK;
  const int tid = threadIdx.x;                    // owns cols 2*tid, 2*tid+1

  const float*  al   = align + (size_t)b * S_ + s0;
  const float2* enc2 = (const float2*)(encs + ((size_t)b * S_ + s0) * E_) + tid;

  float ax = 0.0f, ay = 0.0f;
  for (int s = 0; s < SCHUNK; ++s) {
    float a  = al[s];                             // uniform -> scalar load
    float2 e = enc2[(size_t)s * (E_ / 2)];        // coalesced b64
    ax = fmaf(a, e.x, ax);
    ay = fmaf(a, e.y, ay);
  }
  float2* dst = (float2*)(part + ((size_t)b * NCHUNK + c) * E_) + tid;
  float2 o; o.x = ax; o.y = ay;
  *dst = o;
}

__global__ __launch_bounds__(256) void k_context_reduce(const float* __restrict__ part,
                                                        float* __restrict__ ctx) {
  int idx = blockIdx.x * 256 + threadIdx.x;       // 0 .. 8191
  int b = idx >> 9;
  int col = idx & (E_ - 1);
  float s = 0.0f;
  for (int c = 0; c < NCHUNK; ++c)
    s += part[((size_t)b * NCHUNK + c) * E_ + col];
  ctx[idx] = s;
}

// ---------------------------------------------------------------------------
extern "C" void kernel_launch(void* const* d_in, const int* in_sizes, int n_in,
                              void* d_out, int out_size, void* d_ws, size_t ws_size,
                              hipStream_t stream) {
  const float* hidden = (const float*)d_in[0];   // [16,512]
  const float* encs   = (const float*)d_in[1];   // [16,8192,512]
  const int*   mstep  = (const int*)  d_in[2];   // [1]
  const float* W1     = (const float*)d_in[3];   // [512,512]
  const float* b1     = (const float*)d_in[4];   // [512]
  const float* W2     = (const float*)d_in[5];   // [512,512]
  const float* b2     = (const float*)d_in[6];   // [512]
  const float* V      = (const float*)d_in[7];   // [512]
  const float* bV     = (const float*)d_in[8];   // [1]
  float* out = (float*)d_out;                    // context[16,512] ++ align[16,8192]

  char* ws = (char*)d_ws;
  bf16_t* W2T    = (bf16_t*)ws;                                  // 512 KB
  float*  qb     = (float*)(ws + 512 * 1024);                    //  32 KB
  float*  scores = (float*)(ws + 512 * 1024 + 32 * 1024);        // 512 KB
  float*  cpart  = (float*)(ws + 512 * 1024 + 32 * 1024 + 512 * 1024); // 1 MB

  k_w2t<<<(E_ * U_) / 256, 256, 0, stream>>>(W2, W2T);
  k_qb <<<(B_ * U_) / 256, 256, 0, stream>>>(hidden, W1, b1, b2, qb);

  size_t smem = (size_t)SROWS * LDSTRIDE * sizeof(bf16_t) + 8 * SROWS * sizeof(float);
  k_scores<<<B_ * (S_ / SROWS), 512, smem, stream>>>(encs, W2T, qb, V, bV, scores);

  k_softmax<<<B_, 1024, 0, stream>>>(scores, mstep, out);

  k_context_part  <<<B_ * NCHUNK, 256, 0, stream>>>(encs, out + B_ * E_, cpart);
  k_context_reduce<<<(B_ * E_) / 256, 256, 0, stream>>>(cpart, out);
}
